// ExpertParallelBehindBlock_47863115546644
// MI455X (gfx1250) — matrix operand order repaired and checked by reading, not compiled, not run on hardware
//
#include <hip/hip_runtime.h>

typedef __attribute__((ext_vector_type(16))) _Float16 v16h;
typedef __attribute__((ext_vector_type(8)))  _Float16 v8h;
typedef __attribute__((ext_vector_type(4)))  _Float16 v4h;
typedef __attribute__((ext_vector_type(8)))  float    v8f;

#define BM 128
#define BN 128
#define BK 32
// LDS row stride in halves: 40 halves = 80 bytes -> every fragment base
// (row*80 + {0,16,32} B) stays 16-byte aligned for ds_load_b128, and the
// non-power-of-2 stride spreads rows across banks.
#define LDSS 40

// D[M,N] = A[M,K] (f32 -> f16) * B[K,N] (+ bias), f32 accumulate via WMMA.
// B_IS_F16: B already f16 (stage 2 reads the ws intermediate).
// OUT_F16 : write f16 (stage-1 intermediate) vs f32 (final output).
template <bool B_IS_F16, bool ADD_BIAS, bool OUT_F16>
__global__ __launch_bounds__(256) void moe_gemm_wmma(
    const float* __restrict__ Abase,
    const void*  __restrict__ Bbase,
    const float* __restrict__ biasBase,
    void*        __restrict__ Obase,
    int M, int N, int K,
    long batA, long batB, long batBias, long batO)
{
    // double-buffered staging tiles: 2 x (10 KB + 10 KB) = 40 KB of 320 KB WGP LDS
    __shared__ __align__(16) _Float16 lds_a[2][BM * LDSS];
    __shared__ __align__(16) _Float16 lds_b[2][BN * LDSS];

    const int tid   = threadIdx.x;
    const int lane  = tid & 31;
    const int wave  = tid >> 5;
    const int wm    = wave >> 2;   // 0..1 -> 64-row slab of the 128x128 tile
    const int wn    = wave & 3;    // 0..3 -> 32-col slab
    const int l16   = lane & 15;
    const int lhalf = lane >> 4;   // 0/1

    const int nb = blockIdx.x * BN;
    const int mb = blockIdx.y * BM;
    const int e  = blockIdx.z;     // expert index (stage 1); 0 in stage 2

    const float*    A  = Abase + (size_t)e * batA;
    const float*    Bf = nullptr;
    const _Float16* Bh = nullptr;
    if constexpr (B_IS_F16) Bh = (const _Float16*)Bbase + (size_t)e * batB;
    else                    Bf = (const float*)Bbase    + (size_t)e * batB;

    v8f acc[4][2] = {};

    // cooperative-load index math (256 threads)
    const int ar = tid >> 1;            // A: row 0..127
    const int ac = (tid & 1) * 16;      //    16-wide k chunk
    const int k0 = (tid >> 5) * 4;      // B: k rows k0..k0+3 (0..28)
    const int nc = (tid & 31) * 4;      //    4-wide col chunk (0..124)

    // staged next-tile registers (already converted to f16)
    v8h haLo, haHi;                     // A: 16 halves, k-contiguous
    v4h hb[4];                          // B: 4 cols x 4 k-contiguous halves

    auto load_stage = [&](int kt) {
        {   // A: 128x32 fp32 tile slice -> 16 halves per thread
            const float4* s4 = (const float4*)(A + (size_t)(mb + ar) * K + kt + ac);
            float4 f0 = s4[0], f1 = s4[1], f2 = s4[2], f3 = s4[3];
            haLo = v8h{(_Float16)f0.x, (_Float16)f0.y, (_Float16)f0.z, (_Float16)f0.w,
                       (_Float16)f1.x, (_Float16)f1.y, (_Float16)f1.z, (_Float16)f1.w};
            haHi = v8h{(_Float16)f2.x, (_Float16)f2.y, (_Float16)f2.z, (_Float16)f2.w,
                       (_Float16)f3.x, (_Float16)f3.y, (_Float16)f3.z, (_Float16)f3.w};
        }
        if constexpr (B_IS_F16) {       // B: 4 k-rows x 4 cols, f16 source
            v4h g0 = *(const v4h*)(Bh + (size_t)(kt + k0 + 0) * N + nb + nc);
            v4h g1 = *(const v4h*)(Bh + (size_t)(kt + k0 + 1) * N + nb + nc);
            v4h g2 = *(const v4h*)(Bh + (size_t)(kt + k0 + 2) * N + nb + nc);
            v4h g3 = *(const v4h*)(Bh + (size_t)(kt + k0 + 3) * N + nb + nc);
            hb[0] = v4h{g0[0], g1[0], g2[0], g3[0]};
            hb[1] = v4h{g0[1], g1[1], g2[1], g3[1]};
            hb[2] = v4h{g0[2], g1[2], g2[2], g3[2]};
            hb[3] = v4h{g0[3], g1[3], g2[3], g3[3]};
        } else {                        // B: 4 k-rows x 4 cols, fp32 source
            float4 f0 = *(const float4*)(Bf + (size_t)(kt + k0 + 0) * N + nb + nc);
            float4 f1 = *(const float4*)(Bf + (size_t)(kt + k0 + 1) * N + nb + nc);
            float4 f2 = *(const float4*)(Bf + (size_t)(kt + k0 + 2) * N + nb + nc);
            float4 f3 = *(const float4*)(Bf + (size_t)(kt + k0 + 3) * N + nb + nc);
            hb[0] = v4h{(_Float16)f0.x, (_Float16)f1.x, (_Float16)f2.x, (_Float16)f3.x};
            hb[1] = v4h{(_Float16)f0.y, (_Float16)f1.y, (_Float16)f2.y, (_Float16)f3.y};
            hb[2] = v4h{(_Float16)f0.z, (_Float16)f1.z, (_Float16)f2.z, (_Float16)f3.z};
            hb[3] = v4h{(_Float16)f0.w, (_Float16)f1.w, (_Float16)f2.w, (_Float16)f3.w};
        }
    };

    auto store_stage = [&](int buf) {
        _Float16* da = &lds_a[buf][ar * LDSS + ac];
        *(v8h*)da       = haLo;         // ds_store_b128
        *(v8h*)(da + 8) = haHi;         // ds_store_b128
#pragma unroll
        for (int c = 0; c < 4; ++c)     // transposed: lds_b[n][k], k-contiguous
            *(v4h*)&lds_b[buf][(nc + c) * LDSS + k0] = hb[c];  // ds_store_b64
    };

    const int nk = K / BK;
    load_stage(0);
    store_stage(0);
    __syncthreads();

    for (int it = 0; it < nk; ++it) {
        const int  buf  = it & 1;
        const bool more = (it + 1) < nk;

        // issue next tile's global loads first (latency overlap with WMMA)
        if (more) {
            if (it + 2 < nk) {
                __builtin_prefetch(A + (size_t)(mb + ar) * K + (it + 2) * BK + ac, 0, 1);
                if constexpr (B_IS_F16)
                    __builtin_prefetch(Bh + (size_t)((it + 2) * BK + k0) * N + nb + nc, 0, 1);
                else
                    __builtin_prefetch(Bf + (size_t)((it + 2) * BK + k0) * N + nb + nc, 0, 1);
            }
            load_stage((it + 1) * BK);
        }

        // ---- fragment loads (ISA 16-bit layouts) + 8 WMMAs per wave
        v16h afrag[4];
#pragma unroll
        for (int i = 0; i < 4; ++i) {
            const _Float16* pa =
                &lds_a[buf][(wm * 64 + i * 16 + l16) * LDSS + lhalf * 8];
            v8h lo = *(const v8h*)pa;           // K = h*8 + 0..7
            v8h hi = *(const v8h*)(pa + 16);    // K = 16 + h*8 + 0..7
            afrag[i] = __builtin_shufflevector(lo, hi,
                0,1,2,3,4,5,6,7,8,9,10,11,12,13,14,15);
        }
        v16h bfrag[2];
#pragma unroll
        for (int j = 0; j < 2; ++j) {
            const _Float16* pb =
                &lds_b[buf][(wn * 32 + j * 16 + l16) * LDSS + lhalf * 8];
            v8h lo = *(const v8h*)pb;           // K = h*8 + 0..7
            v8h hi = *(const v8h*)(pb + 16);    // K = 16 + h*8 + 0..7
            bfrag[j] = __builtin_shufflevector(lo, hi,
                0,1,2,3,4,5,6,7,8,9,10,11,12,13,14,15);
        }
#pragma unroll
        for (int i = 0; i < 4; ++i)
#pragma unroll
            for (int j = 0; j < 2; ++j)
                acc[i][j] = __builtin_amdgcn_wmma_f32_16x16x32_f16(
                    false, afrag[i], false, bfrag[j],
                    (short)0, acc[i][j], false, false);

        // stash next tile into the other LDS buffer (no conflict with reads)
        if (more) store_stage(buf ^ 1);
        __syncthreads();
    }

    // ---- epilogue: optional bias, store f16 (stage 1) or f32 (stage 2)
    _Float16* OutH = nullptr;
    float*    OutF = nullptr;
    if constexpr (OUT_F16) OutH = (_Float16*)Obase + (size_t)e * batO;
    else                   OutF = (float*)Obase    + (size_t)e * batO;

#pragma unroll
    for (int j = 0; j < 2; ++j) {
        const int col = nb + wn * 32 + j * 16 + l16;
        float bv = 0.0f;
        if constexpr (ADD_BIAS) bv = biasBase[(size_t)e * batBias + col];
#pragma unroll
        for (int i = 0; i < 4; ++i) {
            const int mrow = mb + wm * 64 + i * 16 + lhalf * 8;
#pragma unroll
            for (int v = 0; v < 8; ++v) {
                const float val = acc[i][j][v] + bv;
                const size_t idx = (size_t)(mrow + v) * N + col;
                if constexpr (OUT_F16) OutH[idx] = (_Float16)val;
                else                   OutF[idx] = val;
            }
        }
    }
}

extern "C" void kernel_launch(void* const* d_in, const int* in_sizes, int n_in,
                              void* d_out, int out_size, void* d_ws, size_t ws_size,
                              hipStream_t stream) {
    (void)in_sizes; (void)n_in; (void)out_size; (void)ws_size;

    const float* expert_input = (const float*)d_in[0]; // [8,1024,4096]
    const float* weight       = (const float*)d_in[1]; // [8,4096,1024]
    const float* bias         = (const float*)d_in[2]; // [8,1,1024]
    const float* combine      = (const float*)d_in[3]; // [8192,8192]
    float*       out          = (float*)d_out;         // [8192,1024]
    _Float16*    eout         = (_Float16*)d_ws;       // [8192,1024] f16, 16 MB

    constexpr int E = 8, C = 1024, DIN = 4096, DOUT = 1024, NTOK = E * C;

    // Stage 1: per-expert GEMM + bias -> f16 intermediate in ws
    moe_gemm_wmma<false, true, true>
        <<<dim3(DOUT / BN, C / BM, E), 256, 0, stream>>>(
            expert_input, weight, bias, eout,
            C, DOUT, DIN,
            (long)C * DIN, (long)DIN * DOUT, (long)DOUT, (long)C * DOUT);

    // Stage 2: combine_weights @ expert_out -> f32 output
    moe_gemm_wmma<true, false, false>
        <<<dim3(DOUT / BN, NTOK / BM, 1), 256, 0, stream>>>(
            combine, eout, nullptr, out,
            NTOK, DOUT, NTOK,
            0, 0, 0, 0);
}
